// VectorQuantizer_47528108098066
// MI455X (gfx1250) — compile-verified, matrix operand order checked by reading, not run on hardware
//
#include <hip/hip_runtime.h>
#include <hip/hip_bf16.h>

// CDNA5 / gfx1250 VectorQuantizer:
//   - TDM (tensor_load_to_lds) double-buffered staging of codebook tiles,
//     with descriptor-driven LDS padding for conflict-free ds_load_b128
//   - WMMA f16->f32 distance GEMM + per-row argmin (shfl_xor reduce)
//   - gather + MSE, WMMA ortho regularizer

typedef __attribute__((ext_vector_type(16))) _Float16 v16h;
typedef __attribute__((ext_vector_type(8)))  _Float16 v8h;
typedef __attribute__((ext_vector_type(8)))  float    v8f;
typedef __attribute__((ext_vector_type(4)))  unsigned int v4u;
typedef __attribute__((ext_vector_type(8)))  int      v8i;
typedef __attribute__((ext_vector_type(4)))  int      v4i;

#define K_CODES 1024
#define DDIM    256
#define HW      1024         // 32*32
#define BSTRIDE 262144       // 256*1024 floats per batch in x[B,C,H,W]

// LDS tile: 16 rows x 256 f16, padded 16B per 512B row by the TDM -> 528B stride
#define ROW_F16   264        // 528 bytes / 2
#define TILE_F16  (16 * ROW_F16)   // 4224 f16 = 8448 B per buffer

union V16U { v16h v; v8h h[2]; };

// ---------------------------------------------------------------------------
// Issue one TDM load: w16 rows [n0, n0+16) -> LDS at lds_off (padded layout).
// D# per CDNA5 ISA ch.8: group0 {count,lds,global,type=2}, group1 {fmt,dims}.
// ---------------------------------------------------------------------------
__device__ __forceinline__ void tdm_issue_tile(const _Float16* __restrict__ w16,
                                               unsigned lds_off, int n0) {
    unsigned long long ga = (unsigned long long)(const void*)(w16 + (long)n0 * DDIM);
    v4u g0;
    g0[0] = 1u;                                        // count=1, no gather
    g0[1] = lds_off;                                   // lds_addr (bytes)
    g0[2] = (unsigned)(ga & 0xFFFFFFFFu);              // global_addr[31:0]
    g0[3] = (unsigned)((ga >> 32) & 0x01FFFFFFu)       // global_addr[56:32]
          | 0x80000000u;                               // type=2 ("image")
    v8i g1;
    g1[0] = (int)((1u << 16)        // data_size = 2 bytes
                | (1u << 20)        // pad_enable
                | (6u << 22)        // pad_interval: 128 dwords (512 B)
                | (3u << 25));      // pad_amount:   4 dwords (16 B)
    g1[1] = (int)(256u << 16);      // tensor_dim0[15:0] = 256 (bits 63:48)
    g1[2] = (int)(1024u << 16);     // tensor_dim1[15:0] = 1024 (bits 95:80)
    g1[3] = (int)(256u << 16);      // tile_dim0 = 256 (bits 127:112)
    g1[4] = 16;                     // tile_dim1 = 16  (bits 143:128)
    g1[5] = 256;                    // tensor_dim0_stride[31:0] = 256 elems
    g1[6] = 0;
    g1[7] = 0;
    v4i z4 = {0, 0, 0, 0};
#if defined(__clang_major__) && __clang_major__ >= 23
    v8i z8 = {0, 0, 0, 0, 0, 0, 0, 0};
    __builtin_amdgcn_tensor_load_to_lds(g0, g1, z4, z4, z8, 0);
#else
    __builtin_amdgcn_tensor_load_to_lds(g0, g1, z4, z4, 0);
#endif
}

// ---------------------------------------------------------------------------
// Prep: per-code norm^2, f16 copy of W, f16 copy of row-normalized W.
// ---------------------------------------------------------------------------
__global__ void vq_prep_w(const float* __restrict__ w,
                          _Float16* __restrict__ w16,
                          _Float16* __restrict__ wn16,
                          float* __restrict__ wnorm) {
    __shared__ float red[256];
    const int k = blockIdx.x;
    const int d = threadIdx.x;
    const float v = w[k * DDIM + d];
    red[d] = v * v;
    __syncthreads();
    for (int s = 128; s > 0; s >>= 1) {
        if (d < s) red[d] += red[d + s];
        __syncthreads();
    }
    const float n2  = red[0];
    const float inv = rsqrtf(n2);
    w16[k * DDIM + d]  = (_Float16)v;
    wn16[k * DDIM + d] = (_Float16)(v * inv);
    if (d == 0) wnorm[k] = n2;
}

// ---------------------------------------------------------------------------
// Distance + argmin. Each wave owns 16 rows of X resident as the A operand.
// Wave 0 TDM-prefetches the next 16-code tile into the alternate LDS buffer
// while all 8 waves run 8x v_wmma_f32_16x16x32_f16 on the current tile read
// via ds_load_b128 from the padded layout. d = ||e||^2 - 2*dot folded with
// one FMA; running (min,argmin) per lane; shfl_xor reduce over column lanes.
// ---------------------------------------------------------------------------
__global__ void vq_argmin(const float* __restrict__ x,
                          const _Float16* __restrict__ w16,
                          const float* __restrict__ wnorm,
                          int* __restrict__ idx_out) {
    __shared__ __align__(16) _Float16 wtile[2][TILE_F16];

    const int lane = threadIdx.x & 31;
    const int wave = threadIdx.x >> 5;
    const int m    = lane & 15;       // A row / B col / C col within tile
    const int hh   = lane >> 4;       // lane half: k sub-slice & C row offset
    const int rowBase = (blockIdx.x * 8 + wave) * 16;
    const int n = rowBase + m;        // this lane's X row
    const long xbase = (long)(n >> 10) * BSTRIDE + (long)(n & (HW - 1));

    const unsigned lds0 = (unsigned)(unsigned long long)(void*)&wtile[0][0];

    // Prologue: start DMA of code tile 0 while we convert A.
    if (wave == 0) tdm_issue_tile(w16, lds0, 0);

    // Load A: 16 rows x 256 ch of x, NCHW-strided, converted to f16.
    V16U a[8];
#pragma unroll
    for (int kk = 0; kk < 8; ++kk) {
        const int d0 = kk * 32 + hh * 16;
#pragma unroll
        for (int e = 0; e < 16; ++e)
            a[kk].v[e] = (_Float16)x[xbase + (long)(d0 + e) * HW];
    }

    float best[8];
    int   bidx[8];
#pragma unroll
    for (int r = 0; r < 8; ++r) { best[r] = 3.4e38f; bidx[r] = 0; }

    for (int t = 0; t < K_CODES / 16; ++t) {
        const int parity = t & 1;
        if (wave == 0) __builtin_amdgcn_s_wait_tensorcnt(0);
        __syncthreads();                       // tile t resident for all waves
        if (wave == 0 && t + 1 < K_CODES / 16) // prefetch tile t+1
            tdm_issue_tile(w16, lds0 + (unsigned)((parity ^ 1) * TILE_F16 * 2),
                           (t + 1) * 16);

        const _Float16* wrow = &wtile[parity][m * ROW_F16 + hh * 16];
        v8f c = {};
#pragma unroll
        for (int kk = 0; kk < 8; ++kk) {
            V16U b;
            b.h[0] = *(const v8h*)(wrow + kk * 32);
            b.h[1] = *(const v8h*)(wrow + kk * 32 + 8);
            c = __builtin_amdgcn_wmma_f32_16x16x32_f16(
                    false, a[kk].v, false, b.v, (short)0, c, false, false);
        }
        const int   col = t * 16 + m;
        const float wn  = wnorm[col];          // this lane's column norm^2
#pragma unroll
        for (int r = 0; r < 8; ++r) {
            const float dist = fmaf(-2.0f, c[r], wn);
            if (dist < best[r]) { best[r] = dist; bidx[r] = col; }
        }
    }

    // Reduce over the 16 columns (lanes) of each half; halves independent.
#pragma unroll
    for (int off = 8; off >= 1; off >>= 1) {
#pragma unroll
        for (int r = 0; r < 8; ++r) {
            const float ob = __shfl_xor(best[r], off, 32);
            const int   oi = __shfl_xor(bidx[r], off, 32);
            if (ob < best[r] || (ob == best[r] && oi < bidx[r])) {
                best[r] = ob; bidx[r] = oi;
            }
        }
    }
    if (m == 0) {                  // lane 0 -> rows 0..7, lane 16 -> rows 8..15
#pragma unroll
        for (int r = 0; r < 8; ++r)
            idx_out[rowBase + r + 8 * hh] = bidx[r];
    }
}

// ---------------------------------------------------------------------------
// Gather codes into NCHW output + accumulate sum((x_q - x)^2).
// ---------------------------------------------------------------------------
__global__ void vq_gather(const float* __restrict__ x,
                          const float* __restrict__ w,
                          const int* __restrict__ idx,
                          float* __restrict__ out,
                          float* __restrict__ accum) {
    __shared__ float red[256];
    const int t  = blockIdx.x * 256 + threadIdx.x;   // flat NCHW index
    const int b  = t >> 18;                          // / (256*1024)
    const int c  = (t >> 10) & 255;
    const int hw = t & 1023;
    const int qi = idx[(b << 10) + hw];
    const float q = w[qi * DDIM + c];
    out[t] = q;
    const float diff = q - x[t];
    red[threadIdx.x] = diff * diff;
    __syncthreads();
    for (int s = 128; s > 0; s >>= 1) {
        if (threadIdx.x < s) red[threadIdx.x] += red[threadIdx.x + s];
        __syncthreads();
    }
    if (threadIdx.x == 0) atomicAdd(accum, red[0]);
}

__global__ void vq_idx_out(const int* __restrict__ idx,
                           float* __restrict__ out_idx) {
    const int t = blockIdx.x * 256 + threadIdx.x;
    out_idx[t] = (float)idx[t];
}

// ---------------------------------------------------------------------------
// Ortho: sum over all (i,j) of (wn_i . wn_j)^2 via WMMA tiles.
// ---------------------------------------------------------------------------
__global__ void vq_ortho(const _Float16* __restrict__ wn16,
                         float* __restrict__ accum) {
    const int lane = threadIdx.x & 31;
    const int wave = threadIdx.x >> 5;
    const int m    = lane & 15;
    const int hh   = lane >> 4;
    const int tile = blockIdx.x * 8 + wave;          // 64x64 tiles of 16x16
    const int i0 = (tile >> 6) * 16;
    const int j0 = (tile & 63) * 16;
    const _Float16* arow = wn16 + (i0 + m) * DDIM + hh * 16;
    const _Float16* brow = wn16 + (j0 + m) * DDIM + hh * 16;
    v8f c = {};
#pragma unroll
    for (int kk = 0; kk < 8; ++kk) {
        V16U a, b;
        a.h[0] = *(const v8h*)(arow + kk * 32);
        a.h[1] = *(const v8h*)(arow + kk * 32 + 8);
        b.h[0] = *(const v8h*)(brow + kk * 32);
        b.h[1] = *(const v8h*)(brow + kk * 32 + 8);
        c = __builtin_amdgcn_wmma_f32_16x16x32_f16(
                false, a.v, false, b.v, (short)0, c, false, false);
    }
    float s = 0.0f;
#pragma unroll
    for (int r = 0; r < 8; ++r) s = fmaf(c[r], c[r], s);
#pragma unroll
    for (int off = 16; off >= 1; off >>= 1)
        s += __shfl_xor(s, off, 32);
    if (lane == 0) atomicAdd(accum, s);
}

// ---------------------------------------------------------------------------
// loss = (1+beta)*mse + 10 * sqrt(sum(S^2) - K) / K^2
// ---------------------------------------------------------------------------
__global__ void vq_finalize(const float* __restrict__ accum,
                            float* __restrict__ loss_out) {
    const float mse  = accum[0] / 16777216.0f;       // N*D elements
    const float orth = accum[1];
    const float fro  = sqrtf(fmaxf(orth - 1024.0f, 0.0f));
    loss_out[0] = 1.25f * mse + 10.0f * (fro / 1048576.0f);
}

// ---------------------------------------------------------------------------
extern "C" void kernel_launch(void* const* d_in, const int* in_sizes, int n_in,
                              void* d_out, int out_size, void* d_ws, size_t ws_size,
                              hipStream_t stream) {
    const float* x = (const float*)d_in[0];   // [64,256,32,32]
    const float* w = (const float*)d_in[1];   // [1024,256]
    float* out = (float*)d_out;               // 4194304 out + 1 loss + 65536 idx

    char* ws = (char*)d_ws;
    _Float16* w16   = (_Float16*)(ws);                  // 512 KB
    _Float16* wn16  = (_Float16*)(ws + 524288);         // 512 KB
    float*    wnorm = (float*)(ws + 1048576);           // 4 KB
    int*      idx   = (int*)(ws + 1052672);             // 256 KB
    float*    accum = (float*)(ws + 1314816);           // 2 floats: mse, ortho

    hipMemsetAsync(accum, 0, 2 * sizeof(float), stream);
    vq_prep_w <<<1024,  256, 0, stream>>>(w, w16, wn16, wnorm);
    vq_argmin <<<512,   256, 0, stream>>>(x, w16, wnorm, idx);
    vq_gather <<<16384, 256, 0, stream>>>(x, w, idx, out, accum);
    vq_idx_out<<<256,   256, 0, stream>>>(idx, out + 4194305);
    vq_ortho  <<<512,   256, 0, stream>>>(wn16, accum + 1);
    vq_finalize<<<1, 1, 0, stream>>>(accum, out + 4194304);
}
